// Contrastive_loss_71966472012554
// MI455X (gfx1250) — matrix operand order repaired and checked by reading, not compiled, not run on hardware
//
#include <hip/hip_runtime.h>
#include <hip/hip_bf16.h>

// ---------------------------------------------------------------------------
// Contrastive loss, B=8192, D=512 fp32.
//   Pass 1 (prep): row norms + one-shot fp32 -> split-bf16 (hi/lo) conversion
//                  into workspace (removes conversion from the GEMM hot loop).
//   Pass 2 (gemm): 128x128 tile per 256-thread block, double-buffered LDS fed
//                  by GLOBAL_LOAD_ASYNC_TO_LDS_B128 (ASYNCcnt) when available,
//                  24x v_wmma_f32_16x16x32_bf16 per K-step (hi*hi+hi*lo+lo*hi),
//                  fused hinge epilogue + wave-reduced atomic scalar.
// ---------------------------------------------------------------------------

typedef __attribute__((ext_vector_type(16))) __bf16 v16bf;
typedef __attribute__((ext_vector_type(8)))  float  v8f;
typedef int v4i __attribute__((vector_size(16)));   // matches builtin's int4

#define B_ROWS 8192
#define D_DIM  512
#define ROW_U32 (D_DIM / 2)        // 256 uints per bf16 row
#define MARGIN_NEG 0.4f
#define MARGIN_POS 0.01f
#define LDS_PITCH_U32 20           // 40 bf16/row (32 data + 8 pad) = 80B, conflict-free

#define GLOBAL_AS __attribute__((address_space(1)))
#define LDS_AS    __attribute__((address_space(3)))

__device__ __forceinline__ unsigned int bf16_rn_bits(float x) {
  unsigned int u = __float_as_uint(x);
  return (u + 0x7FFFu + ((u >> 16) & 1u)) >> 16;   // round-to-nearest-even
}

// Split float4 -> packed bf16 hi (uint2) and bf16 residual lo (uint2).
__device__ __forceinline__ void split4(float4 v, uint2& hi, uint2& lo) {
  unsigned int b0 = bf16_rn_bits(v.x), b1 = bf16_rn_bits(v.y);
  unsigned int b2 = bf16_rn_bits(v.z), b3 = bf16_rn_bits(v.w);
  float l0 = v.x - __uint_as_float(b0 << 16);
  float l1 = v.y - __uint_as_float(b1 << 16);
  float l2 = v.z - __uint_as_float(b2 << 16);
  float l3 = v.w - __uint_as_float(b3 << 16);
  hi.x = b0 | (b1 << 16);
  hi.y = b2 | (b3 << 16);
  lo.x = bf16_rn_bits(l0) | (bf16_rn_bits(l1) << 16);
  lo.y = bf16_rn_bits(l2) | (bf16_rn_bits(l3) << 16);
}

// 16B global -> LDS copy: async DMA (no VGPR staging) when the toolchain has
// the gfx1250 builtin, otherwise plain load/store.
__device__ __forceinline__ void copy16_g2l(const unsigned int* gsrc, unsigned int* ldst) {
#if __has_builtin(__builtin_amdgcn_global_load_async_to_lds_b128)
  __builtin_amdgcn_global_load_async_to_lds_b128(
      (GLOBAL_AS v4i*)gsrc, (LDS_AS v4i*)ldst, 0, 0);
#else
  *reinterpret_cast<int4*>(ldst) = *reinterpret_cast<const int4*>(gsrc);
#endif
}

__device__ __forceinline__ void wait_g2l() {
#if __has_builtin(__builtin_amdgcn_global_load_async_to_lds_b128)
#if __has_builtin(__builtin_amdgcn_s_wait_asynccnt)
  __builtin_amdgcn_s_wait_asynccnt(0);
#else
  asm volatile("s_wait_asynccnt 0" ::: "memory");
#endif
#endif
  // fallback path: plain LDS stores; compiler inserts dscnt waits at barrier
}

// A-fragment (16x32 bf16): lane<16 holds K0..15 of row lt,
// lane>=16 holds K16..31.
__device__ __forceinline__ v16bf load_fragA(const unsigned int* rowBase, int lh) {
  union { int4 q[2]; v16bf v; } u;
  const int4* p = reinterpret_cast<const int4*>(rowBase + lh * 8);
  u.q[0] = p[0];
  u.q[1] = p[1];
  return u.v;
}

// B-fragment (32x16 bf16): lane<16 = column lt K0..15, lane>=16 = K16..31.
__device__ __forceinline__ v16bf load_fragB(const unsigned int* colBase, int lh) {
  union { int4 q[2]; v16bf v; } u;
  const int4* p = reinterpret_cast<const int4*>(colBase + lh * 8);
  u.q[0] = p[0];
  u.q[1] = p[1];
  return u.v;
}

__global__ void zero_out_kernel(float* out) { out[0] = 0.0f; }

// One wave per row: row norms + split-bf16 conversion written once.
__global__ __launch_bounds__(256) void prep_kernel(
    const float* __restrict__ z1, const float* __restrict__ z2,
    float* __restrict__ sq1, float* __restrict__ sq2,
    unsigned int* __restrict__ z1hi, unsigned int* __restrict__ z1lo,
    unsigned int* __restrict__ z2hi, unsigned int* __restrict__ z2lo) {
  int gtid = blockIdx.x * 256 + threadIdx.x;
  int row  = gtid >> 5;
  int lane = threadIdx.x & 31;
  const float4* r1 = reinterpret_cast<const float4*>(z1 + (size_t)row * D_DIM);
  const float4* r2 = reinterpret_cast<const float4*>(z2 + (size_t)row * D_DIM);
  uint2* p1h = reinterpret_cast<uint2*>(z1hi) + (size_t)row * (D_DIM / 4);
  uint2* p1l = reinterpret_cast<uint2*>(z1lo) + (size_t)row * (D_DIM / 4);
  uint2* p2h = reinterpret_cast<uint2*>(z2hi) + (size_t)row * (D_DIM / 4);
  uint2* p2l = reinterpret_cast<uint2*>(z2lo) + (size_t)row * (D_DIM / 4);
  float s1 = 0.f, s2 = 0.f;
#pragma unroll
  for (int i = 0; i < 4; ++i) {
    int idx = lane + i * 32;
    float4 a = r1[idx];
    float4 b = r2[idx];
    s1 += a.x * a.x + a.y * a.y + a.z * a.z + a.w * a.w;
    s2 += b.x * b.x + b.y * b.y + b.z * b.z + b.w * b.w;
    uint2 hi, lo;
    split4(a, hi, lo);  p1h[idx] = hi;  p1l[idx] = lo;
    split4(b, hi, lo);  p2h[idx] = hi;  p2l[idx] = lo;
  }
#pragma unroll
  for (int off = 16; off > 0; off >>= 1) {
    s1 += __shfl_xor(s1, off, 32);
    s2 += __shfl_xor(s2, off, 32);
  }
  if (lane == 0) { sq1[row] = s1; sq2[row] = s2; }
}

__global__ __launch_bounds__(256) void loss_gemm_kernel(
    const unsigned int* __restrict__ z1hi, const unsigned int* __restrict__ z1lo,
    const unsigned int* __restrict__ z2hi, const unsigned int* __restrict__ z2lo,
    const float* __restrict__ sq1, const float* __restrict__ sq2,
    float* __restrict__ out) {
  __shared__ unsigned int sAhi[2][128 * LDS_PITCH_U32];
  __shared__ unsigned int sAlo[2][128 * LDS_PITCH_U32];
  __shared__ unsigned int sBhi[2][128 * LDS_PITCH_U32];
  __shared__ unsigned int sBlo[2][128 * LDS_PITCH_U32];

  const int tid   = threadIdx.x;
  const int lane  = tid & 31;
  const int wave  = tid >> 5;
  const int waveM = wave & 3;      // 4 M-slots of 32 rows
  const int waveN = wave >> 2;     // 2 N-slots of 64 cols
  const int lt    = lane & 15;
  const int lh    = lane >> 4;

  const int mBase = blockIdx.y * 128;
  const int nBase = blockIdx.x * 128;

  // Tile-copy mapping: 256 threads x 2 passes x 4 arrays, 16B chunks.
  const int cr = tid >> 2;         // 0..63 row within pass
  const int cc = tid & 3;          // 16B chunk within 64B bf16 row segment

  v8f acc[2][4];
#pragma unroll
  for (int mi = 0; mi < 2; ++mi)
#pragma unroll
    for (int ni = 0; ni < 4; ++ni)
      acc[mi][ni] = v8f{};

  auto issue_copy = [&](int buf, int k0) {
    const int gcol = (k0 >> 1) + cc * 4;              // uint offset in row
#pragma unroll
    for (int p = 0; p < 2; ++p) {
      int r    = p * 64 + cr;
      int gA   = (mBase + r) * ROW_U32 + gcol;
      int gB   = (nBase + r) * ROW_U32 + gcol;
      int l    = r * LDS_PITCH_U32 + cc * 4;
      copy16_g2l(z1hi + gA, &sAhi[buf][l]);
      copy16_g2l(z1lo + gA, &sAlo[buf][l]);
      copy16_g2l(z2hi + gB, &sBhi[buf][l]);
      copy16_g2l(z2lo + gB, &sBlo[buf][l]);
    }
  };

  issue_copy(0, 0);

  for (int s = 0; s < D_DIM / 32; ++s) {
    const int cur = s & 1;
    wait_g2l();            // this wave's copies for buf[cur] complete
    __syncthreads();       // all waves' copies visible; prev reads of buf[cur^1] done
    if (s + 1 < D_DIM / 32) issue_copy(cur ^ 1, (s + 1) * 32);

    v16bf Ah[2], Al[2], Bh[4], Bl[4];
#pragma unroll
    for (int mi = 0; mi < 2; ++mi) {
      int r = waveM * 32 + mi * 16 + lt;
      Ah[mi] = load_fragA(&sAhi[cur][r * LDS_PITCH_U32], lh);
      Al[mi] = load_fragA(&sAlo[cur][r * LDS_PITCH_U32], lh);
    }
#pragma unroll
    for (int ni = 0; ni < 4; ++ni) {
      int c = waveN * 64 + ni * 16 + lt;
      Bh[ni] = load_fragB(&sBhi[cur][c * LDS_PITCH_U32], lh);
      Bl[ni] = load_fragB(&sBlo[cur][c * LDS_PITCH_U32], lh);
    }

#pragma unroll
    for (int mi = 0; mi < 2; ++mi)
#pragma unroll
      for (int ni = 0; ni < 4; ++ni) {
        v8f c = acc[mi][ni];
        // cross ~= hi*hi + hi*lo + lo*hi  (split-bf16 fp32 emulation)
        c = __builtin_amdgcn_wmma_f32_16x16x32_bf16(false, Ah[mi], false, Bh[ni],
                                                    (short)0, c, false, false);
        c = __builtin_amdgcn_wmma_f32_16x16x32_bf16(false, Ah[mi], false, Bl[ni],
                                                    (short)0, c, false, false);
        c = __builtin_amdgcn_wmma_f32_16x16x32_bf16(false, Al[mi], false, Bh[ni],
                                                    (short)0, c, false, false);
        acc[mi][ni] = c;
      }
  }

  // Fused hinge epilogue. C/D layout: VGPR j, lanes 0-15 -> (M=j, N=lt),
  // lanes 16-31 -> (M=j+8, N=lt).
  const float invD = 1.0f / (float)D_DIM;
  float psum = 0.f, nsum = 0.f;
  float s2v[4];
  int   nIdx[4];
#pragma unroll
  for (int ni = 0; ni < 4; ++ni) {
    nIdx[ni] = nBase + waveN * 64 + ni * 16 + lt;
    s2v[ni]  = sq2[nIdx[ni]];
  }
#pragma unroll
  for (int mi = 0; mi < 2; ++mi) {
#pragma unroll
    for (int j = 0; j < 8; ++j) {
      int   m  = mBase + waveM * 32 + mi * 16 + lh * 8 + j;
      float s1 = sq1[m];
#pragma unroll
      for (int ni = 0; ni < 4; ++ni) {
        float dist = (s1 + s2v[ni] - 2.0f * acc[mi][ni][j]) * invD;
        if (m == nIdx[ni])
          psum += fmaxf(dist - MARGIN_POS, 0.0f);
        else
          nsum += fmaxf(MARGIN_NEG - dist, 0.0f);
      }
    }
  }
  float contrib = (1.5f / (float)B_ROWS) * psum +
                  (0.5f / ((float)B_ROWS * (float)(B_ROWS - 1))) * nsum;
#pragma unroll
  for (int off = 16; off > 0; off >>= 1) contrib += __shfl_xor(contrib, off, 32);
  if (lane == 0) unsafeAtomicAdd(out, contrib);
}

extern "C" void kernel_launch(void* const* d_in, const int* in_sizes, int n_in,
                              void* d_out, int out_size, void* d_ws, size_t ws_size,
                              hipStream_t stream) {
  const float* z1 = (const float*)d_in[0];
  const float* z2 = (const float*)d_in[1];
  float* out = (float*)d_out;

  // Workspace layout: sq1|sq2 (64KB) then 4 bf16 matrices (8MB each).
  float* sq1 = (float*)d_ws;
  float* sq2 = sq1 + B_ROWS;
  unsigned int* z1hi = (unsigned int*)(sq2 + B_ROWS);
  unsigned int* z1lo = z1hi + (size_t)B_ROWS * ROW_U32;
  unsigned int* z2hi = z1lo + (size_t)B_ROWS * ROW_U32;
  unsigned int* z2lo = z2hi + (size_t)B_ROWS * ROW_U32;

  zero_out_kernel<<<1, 1, 0, stream>>>(out);
  prep_kernel<<<(B_ROWS * 32) / 256, 256, 0, stream>>>(z1, z2, sq1, sq2,
                                                       z1hi, z1lo, z2hi, z2lo);
  dim3 grid(B_ROWS / 128, B_ROWS / 128);   // 64 x 64 tiles of 128x128
  loss_gemm_kernel<<<grid, 256, 0, stream>>>(z1hi, z1lo, z2hi, z2lo,
                                             sq1, sq2, out);
}
